// NeuronBasedQKV_300647710815
// MI455X (gfx1250) — matrix operand order repaired and checked by reading, not compiled, not run on hardware
//
#include <hip/hip_runtime.h>

// ============================================================================
// NeuronBasedQKV for MI455X (gfx1250).
// All heavy math on v_wmma_f32_16x16x32_bf16 (fp32 accumulation).
// Operands pre-swizzled into fragment-major global layouts; GEMM inner loops
// are {global_load_b128 (B) + register-resident A + v_wmma} with explicit
// kbo/kbi tiling so the compiler cannot hoist/spill the A fragments.
// Workspace requirement: ~132 MB.
// ============================================================================

typedef __attribute__((ext_vector_type(16))) __bf16 v16bf;
typedef __attribute__((ext_vector_type(8)))  float  v8f;

#define DEVI __device__ __forceinline__

namespace {
constexpr int kB    = 4;
constexpr int kS    = 1024;
constexpr int kD    = 1024;
constexpr int kNN   = 256;   // n_neurons
constexpr int kNB   = 32;    // n_basis
constexpr int kRank = 512;
constexpr int kTopK = 8;
constexpr int kT    = kB * kS;  // 4096 tokens
constexpr float kNegInf = -1e30f;

DEVI v8f vzero8() { v8f z; for (int i = 0; i < 8; ++i) z[i] = 0.f; return z; }

// Fragment coordinate maps (wave32 bf16 WMMA 16x16x32), contiguous k order used for both A and B.
// A (16x32, MxK): lane = m + 16*(k>>4), elem = k & 15
DEVI int a_lane(int m, int k) { return m + 16 * (k >> 4); }
DEVI int a_elem(int k)        { return k & 15; }
// B (32x16, KxN): lane = n + 16*(k>>4), elem = k & 15
DEVI int b_lane(int k, int n) { return n + 16 * (k >> 4); }
DEVI int b_elem(int k)        { return k & 15; }
// C/D (16x16 f32): col = lane&15, row = r + 8*(lane>>4)

DEVI v8f wmma_bf16(v16bf a, v16bf b, v8f c) {
  return __builtin_amdgcn_wmma_f32_16x16x32_bf16(false, a, false, b,
                                                 (short)0, c, false, false);
}

// ---- fragment-major global layouts -----------------------------------------
// xfrag   : [rowtile(64)][mblk(4)][kb(32)][lane(32)][e(16)]       (A frags)
// bqk/bvo : [n(32)][coltile(8)][kb(32)][nsub(4)][lane(32)][e(16)] (B frags)
// bvoT    : [n(32)][dtile(16)][kb(16)][nsub(4)][lane(32)][e(16)]  (B frags, transposed)
// Qfrag   : [bh(64)][qtile(16)][mblk(4)][lane(32)][e(16)]         (A frags)
// Kfrag   : [bh(64)][kvtile(32)][kvhalf(2)][lane(32)][e(16)]      (B frags of K^T)
// Vfrag   : [bh(64)][kvtile(32)][dhalf(2)][lane(32)][e(16)]       (B frags of V)
// Ofrag   : [rowtile(64)][mblk(4)][kb(16)][lane(32)][e(16)]       (A frags)

DEVI size_t qfrag_pos(int t, int c) {
  int b = t >> 10, s = t & 1023, h = c >> 5, d = c & 31;
  int bh = b * 16 + h;
  return (size_t)(((bh * 16 + (s >> 6)) * 4 + ((s >> 4) & 3))) * 512
         + a_lane(s & 15, d) * 16 + a_elem(d);
}
DEVI size_t kfrag_pos(int t, int c) {
  int b = t >> 10, s = t & 1023, h = c >> 5, d = c & 31;
  int bh = b * 16 + h, kvt = s >> 5, kv = s & 31;
  return (size_t)((bh * 32 + kvt) * 2 + (kv >> 4)) * 512
         + ((kv & 15) + 16 * (d >> 4)) * 16 + (d & 15);
}
DEVI size_t vfrag_pos(int t, int c) {
  int b = t >> 10, s = t & 1023, h = c >> 5, d = c & 31;
  int bh = b * 16 + h, kvt = s >> 5, kv = s & 31;
  return (size_t)((bh * 32 + kvt) * 2 + (d >> 4)) * 512
         + ((d & 15) + 16 * (kv >> 4)) * 16 + (kv & 15);
}
DEVI size_t ofrag_pos(int t, int c) {
  int rt = t >> 6, mblk = (t >> 4) & 3, m = t & 15;
  int kb = c >> 5, kl = c & 31;
  return (size_t)((rt * 4 + mblk) * 16 + kb) * 512
         + a_lane(m, kl) * 16 + a_elem(kl);
}
} // namespace

// ---------------------------------------------------------------------------
// Pre-swizzle passes (fp32 -> bf16 + fragment layout). Reads coalesced.
// ---------------------------------------------------------------------------
__global__ void swz_x_kernel(const float* __restrict__ x,
                             __bf16* __restrict__ xf, int n) {
  int idx = blockIdx.x * 256 + threadIdx.x;
  if (idx >= n) return;
  int c = idx & 1023, row = idx >> 10;
  int rt = row >> 6, mblk = (row >> 4) & 3, m = row & 15;
  int kb = c >> 5, kl = c & 31;
  size_t pos = (size_t)((rt * 4 + mblk) * 32 + kb) * 512
             + a_lane(m, kl) * 16 + a_elem(kl);
  xf[pos] = (__bf16)x[idx];
}

__global__ void swz_basis_kernel(const float* __restrict__ bsrc,
                                 __bf16* __restrict__ bf, int n) {
  // src: [n][d(1024)][col(512)] -> B frags of (X @ basis)
  int idx = blockIdx.x * 256 + threadIdx.x;
  if (idx >= n) return;
  int col = idx & 511, d = (idx >> 9) & 1023, nb = idx >> 19;
  int ct = col >> 6, nsub = (col >> 4) & 3, nl = col & 15;
  int kb = d >> 5, kl = d & 31;
  size_t pos = (size_t)(((nb * 8 + ct) * 32 + kb) * 4 + nsub) * 512
             + b_lane(kl, nl) * 16 + b_elem(kl);
  bf[pos] = (__bf16)bsrc[idx];
}

__global__ void swz_basisT_kernel(const float* __restrict__ bsrc,
                                  __bf16* __restrict__ bf, int n) {
  // src: [n][d(1024)][r(512)] -> B frags of (O @ basis^T): B[k=r][n=d]
  int idx = blockIdx.x * 256 + threadIdx.x;
  if (idx >= n) return;
  int r = idx & 511, d = (idx >> 9) & 1023, nb = idx >> 19;
  int dt = d >> 6, nsub = (d >> 4) & 3, nl = d & 15;
  int kb = r >> 5, kl = r & 31;
  size_t pos = (size_t)(((nb * 16 + dt) * 16 + kb) * 4 + nsub) * 512
             + b_lane(kl, nl) * 16 + b_elem(kl);
  bf[pos] = (__bf16)bsrc[idx];
}

// ---------------------------------------------------------------------------
// Router: scores = x @ W_router.  16 tokens per block to reuse W columns.
// ---------------------------------------------------------------------------
__global__ void __launch_bounds__(256)
router_kernel(const float* __restrict__ x, const float* __restrict__ W,
              float* __restrict__ scores) {
  __shared__ float xs[16 * 1024];
  const int t0 = blockIdx.x * 16;
  for (int idx = threadIdx.x; idx < 16 * 1024; idx += 256)
    xs[idx] = x[(size_t)(t0 + (idx >> 10)) * kD + (idx & 1023)];
  __syncthreads();
  float acc[16];
  for (int i = 0; i < 16; ++i) acc[i] = 0.f;
  const int nn = threadIdx.x;
  for (int d = 0; d < kD; ++d) {
    float w = W[(size_t)d * kNN + nn];
    for (int i = 0; i < 16; ++i) acc[i] += xs[i * 1024 + d] * w;
  }
  for (int i = 0; i < 16; ++i)
    scores[(size_t)(t0 + i) * kNN + nn] = acc[i];
}

// ---------------------------------------------------------------------------
// Per-token top-8 + routing softmax + recipe mix + basis softmax (1 wave/token)
// ---------------------------------------------------------------------------
__global__ void __launch_bounds__(32)
topk_recipe_kernel(const float* __restrict__ scores,
                   const float* __restrict__ rQ, const float* __restrict__ rK,
                   const float* __restrict__ rV, const float* __restrict__ rO,
                   float* __restrict__ trQ, float* __restrict__ trK,
                   float* __restrict__ trV, float* __restrict__ trO) {
  const int t = blockIdx.x;
  const int lane = threadIdx.x;

  float v[8];
  for (int j = 0; j < 8; ++j) v[j] = scores[(size_t)t * kNN + lane + 32 * j];

  float topv[kTopK]; int topi[kTopK];
  for (int it = 0; it < kTopK; ++it) {
    float bm = v[0]; int bj = 0;
    for (int j = 1; j < 8; ++j) if (v[j] > bm) { bm = v[j]; bj = j; }
    int bn = lane + 32 * bj;
    for (int m = 16; m >= 1; m >>= 1) {
      float om = __shfl_xor(bm, m, 32);
      int   on = __shfl_xor(bn, m, 32);
      if (om > bm || (om == bm && on < bn)) { bm = om; bn = on; }
    }
    topv[it] = bm; topi[it] = bn;
    if ((bn & 31) == lane) v[bn >> 5] = kNegInf;
  }

  float m = topv[0];
  for (int j = 1; j < kTopK; ++j) m = fmaxf(m, topv[j]);
  float w[kTopK], ws = 0.f;
  for (int j = 0; j < kTopK; ++j) { w[j] = __expf(topv[j] - m); ws += w[j]; }
  float inv = 1.f / ws;
  for (int j = 0; j < kTopK; ++j) w[j] *= inv;

  const float* recs[4] = { rQ, rK, rV, rO };
  float* outs[4] = { trQ, trK, trV, trO };
  for (int rr = 0; rr < 4; ++rr) {
    float s = 0.f;
    for (int j = 0; j < kTopK; ++j)
      s += w[j] * recs[rr][(size_t)topi[j] * kNB + lane];
    float mm = s;
    for (int msk = 16; msk >= 1; msk >>= 1)
      mm = fmaxf(mm, __shfl_xor(mm, msk, 32));
    float e = __expf(s - mm);
    float ss = e;
    for (int msk = 16; msk >= 1; msk >>= 1) ss += __shfl_xor(ss, msk, 32);
    outs[rr][(size_t)t * kNB + lane] = e / ss;
  }
}

// ---------------------------------------------------------------------------
// QKV projection. z=0: basis_qk -> Q (tr_Q) and K (tr_K); z=1: basis_vo -> V.
// kbo/kbi tiling keeps exactly 16 A fragments (128 VGPRs) register-resident;
// acc = sum_kbo sum_n tr[n] * (partial GEMM over 8 k-steps).
// ---------------------------------------------------------------------------
__global__ void __launch_bounds__(256)
qkv_proj_kernel(const __bf16* __restrict__ xf,
                const __bf16* __restrict__ bqkf,
                const __bf16* __restrict__ bvof,
                const float* __restrict__ trQ, const float* __restrict__ trK,
                const float* __restrict__ trV,
                __bf16* __restrict__ Qf, __bf16* __restrict__ Kfr,
                __bf16* __restrict__ Vfr) {
  const int mode = blockIdx.z;
  const __bf16* basis = mode ? bvof : bqkf;
  const float*  trA   = mode ? trV : trQ;
  const float*  trB   = trK;

  __shared__ __align__(32) __bf16 Xs[4 * 32 * 512];  // 128 KB, A-frag major
  __shared__ float trAsT[kNB * 64];  // [n][row], rows contiguous for v8f reads
  __shared__ float trBsT[kNB * 64];

  const int tid = threadIdx.x, lane = tid & 31, wave = tid >> 5;
  const int row0 = blockIdx.x * 64;
  const int ct = blockIdx.y;  // 64-col tile

  {
    const uint4* src = (const uint4*)(xf + (size_t)blockIdx.x * (4 * 32 * 512));
    uint4* dst = (uint4*)Xs;
    for (int i = tid; i < 8192; i += 256) dst[i] = src[i];
  }
  for (int idx = tid; idx < 64 * 32; idx += 256) {
    int n = idx >> 6, row = idx & 63;
    trAsT[n * 64 + row] = trA[(size_t)(row0 + row) * kNB + n];
    trBsT[n * 64 + row] = trB[(size_t)(row0 + row) * kNB + n];
  }
  __syncthreads();

  const int n_blk = wave & 3;
  const int m0 = (wave >> 2) * 2, m1 = m0 + 1;
  const int hi = lane >> 4;

  v8f accA0 = vzero8(), accA1 = vzero8();
  v8f accB0 = vzero8(), accB1 = vzero8();

  for (int kbo = 0; kbo < 4; ++kbo) {
    // preload this chunk's A fragments into registers (16 x v16bf = 128 VGPRs)
    v16bf A0[8], A1[8];
#pragma unroll
    for (int i = 0; i < 8; ++i) {
      A0[i] = *(const v16bf*)(Xs + (m0 * 32 + kbo * 8 + i) * 512 + lane * 16);
      A1[i] = *(const v16bf*)(Xs + (m1 * 32 + kbo * 8 + i) * 512 + lane * 16);
    }
#pragma clang loop unroll(disable)
    for (int n = 0; n < kNB; ++n) {
      const __bf16* bptr = basis
          + ((size_t)((n * 8 + ct) * 32 + kbo * 8) * 4 + n_blk) * 512
          + lane * 16;
      v8f P0 = vzero8(), P1 = vzero8();
#pragma unroll
      for (int i = 0; i < 8; ++i) {
        v16bf bfr = *(const v16bf*)(bptr + (size_t)i * 2048);
        P0 = wmma_bf16(A0[i], bfr, P0);
        P1 = wmma_bf16(A1[i], bfr, P1);
      }
      v8f tA0 = *(const v8f*)(trAsT + n * 64 + m0 * 16 + 8 * hi);
      v8f tA1 = *(const v8f*)(trAsT + n * 64 + m1 * 16 + 8 * hi);
      for (int r = 0; r < 8; ++r) {
        accA0[r] += tA0[r] * P0[r];
        accA1[r] += tA1[r] * P1[r];
      }
      if (mode == 0) {
        v8f tB0 = *(const v8f*)(trBsT + n * 64 + m0 * 16 + 8 * hi);
        v8f tB1 = *(const v8f*)(trBsT + n * 64 + m1 * 16 + 8 * hi);
        for (int r = 0; r < 8; ++r) {
          accB0[r] += tB0[r] * P0[r];
          accB1[r] += tB1[r] * P1[r];
        }
      }
    }
  }

  const int colw = ct * 64 + n_blk * 16 + (lane & 15);
  for (int r = 0; r < 8; ++r) {
    int t0 = row0 + m0 * 16 + r + 8 * hi;
    int t1 = row0 + m1 * 16 + r + 8 * hi;
    if (mode == 0) {
      Qf [qfrag_pos(t0, colw)] = (__bf16)accA0[r];
      Qf [qfrag_pos(t1, colw)] = (__bf16)accA1[r];
      Kfr[kfrag_pos(t0, colw)] = (__bf16)accB0[r];
      Kfr[kfrag_pos(t1, colw)] = (__bf16)accB1[r];
    } else {
      Vfr[vfrag_pos(t0, colw)] = (__bf16)accA0[r];
      Vfr[vfrag_pos(t1, colw)] = (__bf16)accA1[r];
    }
  }
}

// ---------------------------------------------------------------------------
// Causal flash attention. Operands already fragment-major: no staging, no
// __syncthreads. Per-wave LDS bounce only for P (D-layout -> A-layout).
// ---------------------------------------------------------------------------
__global__ void __launch_bounds__(128)
attention_kernel(const __bf16* __restrict__ Qf, const __bf16* __restrict__ Kfr,
                 const __bf16* __restrict__ Vfr, __bf16* __restrict__ Ofr) {
  const int qt = blockIdx.x, bh = blockIdx.y;
  const int tid = threadIdx.x, lane = tid & 31, wave = tid >> 5;
  const int q0 = qt * 64;
  const int hi = lane >> 4;
  const int b = bh >> 4, h = bh & 15;
  const float scale = 0.17677669529663687f;  // 1/sqrt(32)

  __shared__ __align__(32) __bf16 Pf[4 * 512];  // per-wave P A-fragment

  v16bf qa = *(const v16bf*)(Qf + (size_t)((bh * 16 + qt) * 4 + wave) * 512
                             + lane * 16);

  v8f o0 = vzero8(), o1 = vzero8();
  float mrow[8], lrow[8];
  for (int r = 0; r < 8; ++r) { mrow[r] = kNegInf; lrow[r] = 0.f; }

  for (int kv0 = 0; kv0 < q0 + 64; kv0 += 32) {
    const size_t kvbase = (size_t)((bh * 32 + (kv0 >> 5)) * 2) * 512;
    v16bf kb0 = *(const v16bf*)(Kfr + kvbase + lane * 16);
    v16bf kb1 = *(const v16bf*)(Kfr + kvbase + 512 + lane * 16);
    v8f s0 = wmma_bf16(qa, kb0, vzero8());
    v8f s1 = wmma_bf16(qa, kb1, vzero8());

    const int colg0 = kv0 + (lane & 15);
    const int colg1 = colg0 + 16;
    for (int r = 0; r < 8; ++r) {
      int rowg = q0 + wave * 16 + r + 8 * hi;
      float a  = s0[r] * scale; if (colg0 > rowg) a = kNegInf;
      float bq = s1[r] * scale; if (colg1 > rowg) bq = kNegInf;
      float mx = fmaxf(a, bq);
      for (int msk = 8; msk >= 1; msk >>= 1)
        mx = fmaxf(mx, __shfl_xor(mx, msk, 32));   // row reduce (16-lane half)
      float mnew = fmaxf(mrow[r], mx);
      float corr = __expf(mrow[r] - mnew);
      float p0 = __expf(a - mnew), p1 = __expf(bq - mnew);
      float rs = p0 + p1;
      for (int msk = 8; msk >= 1; msk >>= 1) rs += __shfl_xor(rs, msk, 32);
      lrow[r] = lrow[r] * corr + rs;
      mrow[r] = mnew;
      o0[r] *= corr; o1[r] *= corr;
      int qrow = r + 8 * hi;
      int c0 = lane & 15, c1 = c0 + 16;
      Pf[wave * 512 + a_lane(qrow, c0) * 16 + a_elem(c0)] = (__bf16)p0;
      Pf[wave * 512 + a_lane(qrow, c1) * 16 + a_elem(c1)] = (__bf16)p1;
    }
    v16bf pa  = *(const v16bf*)(Pf + wave * 512 + lane * 16);
    v16bf vb0 = *(const v16bf*)(Vfr + kvbase + lane * 16);
    v16bf vb1 = *(const v16bf*)(Vfr + kvbase + 512 + lane * 16);
    o0 = wmma_bf16(pa, vb0, o0);
    o1 = wmma_bf16(pa, vb1, o1);
  }

  for (int r = 0; r < 8; ++r) {
    float inv = 1.f / lrow[r];
    int t = b * kS + q0 + wave * 16 + r + 8 * hi;
    int c0 = h * 32 + (lane & 15);
    Ofr[ofrag_pos(t, c0)]      = (__bf16)(o0[r] * inv);
    Ofr[ofrag_pos(t, c0 + 16)] = (__bf16)(o1[r] * inv);
  }
}

// ---------------------------------------------------------------------------
// Output projection: out[t,d] = sum_n trO[t,n] * (O[t,:] @ basis_vo[n]^T)[d]
// Same kbo/kbi tiling as the QKV projection.
// ---------------------------------------------------------------------------
__global__ void __launch_bounds__(256)
oproj_kernel(const __bf16* __restrict__ Ofr, const __bf16* __restrict__ bvoT,
             const float* __restrict__ trO, float* __restrict__ out) {
  __shared__ __align__(32) __bf16 Os[4 * 16 * 512];  // 64 KB, A-frag major
  __shared__ float trOsT[kNB * 64];

  const int tid = threadIdx.x, lane = tid & 31, wave = tid >> 5;
  const int row0 = blockIdx.x * 64;
  const int dt = blockIdx.y;  // 64-wide d tile

  {
    const uint4* src = (const uint4*)(Ofr + (size_t)blockIdx.x * (4 * 16 * 512));
    uint4* dst = (uint4*)Os;
    for (int i = tid; i < 4096; i += 256) dst[i] = src[i];
  }
  for (int idx = tid; idx < 64 * 32; idx += 256) {
    int n = idx >> 6, row = idx & 63;
    trOsT[n * 64 + row] = trO[(size_t)(row0 + row) * kNB + n];
  }
  __syncthreads();

  const int n_blk = wave & 3;
  const int m0 = (wave >> 2) * 2, m1 = m0 + 1;
  const int hi = lane >> 4;

  v8f acc0 = vzero8(), acc1 = vzero8();

  for (int kbo = 0; kbo < 2; ++kbo) {
    v16bf A0[8], A1[8];
#pragma unroll
    for (int i = 0; i < 8; ++i) {
      A0[i] = *(const v16bf*)(Os + (m0 * 16 + kbo * 8 + i) * 512 + lane * 16);
      A1[i] = *(const v16bf*)(Os + (m1 * 16 + kbo * 8 + i) * 512 + lane * 16);
    }
#pragma clang loop unroll(disable)
    for (int n = 0; n < kNB; ++n) {
      const __bf16* bptr = bvoT
          + ((size_t)((n * 16 + dt) * 16 + kbo * 8) * 4 + n_blk) * 512
          + lane * 16;
      v8f P0 = vzero8(), P1 = vzero8();
#pragma unroll
      for (int i = 0; i < 8; ++i) {
        v16bf bfr = *(const v16bf*)(bptr + (size_t)i * 2048);
        P0 = wmma_bf16(A0[i], bfr, P0);
        P1 = wmma_bf16(A1[i], bfr, P1);
      }
      v8f t0 = *(const v8f*)(trOsT + n * 64 + m0 * 16 + 8 * hi);
      v8f t1 = *(const v8f*)(trOsT + n * 64 + m1 * 16 + 8 * hi);
      for (int r = 0; r < 8; ++r) {
        acc0[r] += t0[r] * P0[r];
        acc1[r] += t1[r] * P1[r];
      }
    }
  }

  const int colw = dt * 64 + n_blk * 16 + (lane & 15);
  for (int r = 0; r < 8; ++r) {
    out[(size_t)(row0 + m0 * 16 + r + 8 * hi) * kD + colw] = acc0[r];
    out[(size_t)(row0 + m1 * 16 + r + 8 * hi) * kD + colw] = acc1[r];
  }
}

// ---------------------------------------------------------------------------
// Host launcher
// ---------------------------------------------------------------------------
extern "C" void kernel_launch(void* const* d_in, const int* in_sizes, int n_in,
                              void* d_out, int out_size, void* d_ws,
                              size_t ws_size, hipStream_t stream) {
  (void)in_sizes; (void)n_in; (void)out_size; (void)ws_size;

  const float* x   = (const float*)d_in[0];
  const float* Wr  = (const float*)d_in[1];
  const float* rQ  = (const float*)d_in[2];
  const float* rK  = (const float*)d_in[3];
  const float* rV  = (const float*)d_in[4];
  const float* rO  = (const float*)d_in[5];
  const float* bqk = (const float*)d_in[6];
  const float* bvo = (const float*)d_in[7];
  // d_in[8] = causal mask, hardcoded in the attention kernel.
  float* out = (float*)d_out;

  size_t off = 0;
  auto take = [&](size_t bytes) -> void* {
    void* p = (char*)d_ws + off;
    off += (bytes + 255) & ~(size_t)255;
    return p;
  };
  __bf16* xfrag   = (__bf16*)take((size_t)kT * kD * 2);
  __bf16* bqkfrag = (__bf16*)take((size_t)kNB * kD * kRank * 2);
  __bf16* bvofrag = (__bf16*)take((size_t)kNB * kD * kRank * 2);
  __bf16* bvoTfrag= (__bf16*)take((size_t)kNB * kD * kRank * 2);
  float*  scores  = (float*) take((size_t)kT * kNN * 4);
  float*  trQ     = (float*) take((size_t)kT * kNB * 4);
  float*  trK     = (float*) take((size_t)kT * kNB * 4);
  float*  trV     = (float*) take((size_t)kT * kNB * 4);
  float*  trO     = (float*) take((size_t)kT * kNB * 4);
  __bf16* Qfrag   = (__bf16*)take((size_t)kT * kRank * 2);
  __bf16* Kfrag   = (__bf16*)take((size_t)kT * kRank * 2);
  __bf16* Vfrag   = (__bf16*)take((size_t)kT * kRank * 2);
  __bf16* Ofrag   = (__bf16*)take((size_t)kT * kRank * 2);

  const int nX = kT * kD;
  const int nB = kNB * kD * kRank;
  swz_x_kernel<<<(nX + 255) / 256, 256, 0, stream>>>(x, xfrag, nX);
  swz_basis_kernel<<<(nB + 255) / 256, 256, 0, stream>>>(bqk, bqkfrag, nB);
  swz_basis_kernel<<<(nB + 255) / 256, 256, 0, stream>>>(bvo, bvofrag, nB);
  swz_basisT_kernel<<<(nB + 255) / 256, 256, 0, stream>>>(bvo, bvoTfrag, nB);

  router_kernel<<<kT / 16, 256, 0, stream>>>(x, Wr, scores);
  topk_recipe_kernel<<<kT, 32, 0, stream>>>(scores, rQ, rK, rV, rO,
                                            trQ, trK, trV, trO);
  qkv_proj_kernel<<<dim3(kT / 64, kRank / 64, 2), 256, 0, stream>>>(
      xfrag, bqkfrag, bvofrag, trQ, trK, trV, Qfrag, Kfrag, Vfrag);
  attention_kernel<<<dim3(kS / 64, kB * 16), 128, 0, stream>>>(
      Qfrag, Kfrag, Vfrag, Ofrag);
  oproj_kernel<<<dim3(kT / 64, kD / 64), 256, 0, stream>>>(
      Ofrag, bvoTfrag, trO, out);
}
